// RelationalGATEncoder_14465449852940
// MI455X (gfx1250) — matrix operand order repaired and checked by reading, not compiled, not run on hardware
//
#include <hip/hip_runtime.h>

// ---------------------------------------------------------------------------
// RelationalGATEncoder for MI455X (gfx1250, wave32).
// Dense projections via v_wmma_f32_16x16x32_bf16 (bf16 operands, f32 accum).
// Edge softmax/aggregation via monotone-uint atomicMax + f32 atomics
// (memory/atomic bound; GEMM is negligible on the WMMA pipe).
// ---------------------------------------------------------------------------

#define NN 100000
#define EE 400000
#define RR 4
#define DD 128
#define CDIV(a, b) (((a) + (b) - 1) / (b))

typedef unsigned short u16;
typedef __attribute__((ext_vector_type(8)))  __bf16 v8bf;
typedef __attribute__((ext_vector_type(16))) __bf16 v16bf;
typedef __attribute__((ext_vector_type(8)))  float  v8f;

union V16u { v16bf v; v8bf h[2]; };

__device__ __forceinline__ u16 f2bf(float f) {
  unsigned u = __float_as_uint(f);
  unsigned r = u + 0x7FFFu + ((u >> 16) & 1u);   // round-to-nearest-even
  return (u16)(r >> 16);
}
// order-preserving float<->uint encoding (for atomicMax on f32 values)
__device__ __forceinline__ unsigned f2mono(float f) {
  unsigned u = __float_as_uint(f);
  return (u & 0x80000000u) ? ~u : (u | 0x80000000u);
}
__device__ __forceinline__ float mono2f(unsigned u) {
  return __uint_as_float((u & 0x80000000u) ? (u & 0x7FFFFFFFu) : ~u);
}

// ---------------------------------------------------------------------------
__global__ void f32_to_bf16_k(const float* __restrict__ s, u16* __restrict__ d, int n) {
  int i = blockIdx.x * 256 + threadIdx.x;
  if (i < n) d[i] = f2bf(s[i]);
}

__global__ void fill_u32_k(unsigned* __restrict__ p, unsigned v, int n) {
  int i = blockIdx.x * 256 + threadIdx.x;
  if (i < n) p[i] = v;
}

// ---------------------------------------------------------------------------
// C[M,128] = A[M,128] @ B[128,128], A/B bf16, C f32.
// Block: 256 threads (8 waves), 64 rows of C. LDS: A tile (16KB) + B^T (32KB).
// Wave w: M-tile (w&3), N-half (w>>2); 4 n-tiles x 4 k-steps = 16 WMMAs.
// Uniform full-tile fast path: only the last of 1563 blocks takes the
// per-element guarded path, so the hot path has zero exec-mask branching.
__global__ __launch_bounds__(256)
void gemm_bf16_k128(const u16* __restrict__ A, const u16* __restrict__ B,
                    float* __restrict__ C, int M) {
  __shared__ __align__(16) u16 sA[64 * 128];
  __shared__ __align__(16) u16 sBt[128 * 128];
  const int m0 = blockIdx.x * 64;
  const bool full = (m0 + 64) <= M;          // uniform (scalar) predicate

  // Stage A tile (coalesced 16B loads; zero-pad out-of-range rows).
  const uint4* Ag = (const uint4*)A;         // 16 uint4 per 128-col row
  uint4* sA4 = (uint4*)sA;
  if (full) {
#pragma unroll
    for (int i = threadIdx.x; i < 64 * 16; i += 256) {
      sA4[i] = Ag[(m0 + (i >> 4)) * 16 + (i & 15)];
    }
  } else {
    for (int i = threadIdx.x; i < 64 * 16; i += 256) {
      uint4 v = {0u, 0u, 0u, 0u};
      int gr = m0 + (i >> 4);
      if (gr < M) v = Ag[gr * 16 + (i & 15)];
      sA4[i] = v;
    }
  }
  // Stage B transposed: sBt[n][k] = B[k][n] (so each lane reads K-contiguous).
  for (int i = threadIdx.x; i < 128 * 128; i += 256) {
    sBt[(i & 127) * 128 + (i >> 7)] = B[i];
  }
  __syncthreads();

  const int lane = threadIdx.x & 31;
  const int wave = threadIdx.x >> 5;
  const int mt   = wave & 3;        // 16-row M tile within block
  const int nh   = wave >> 2;       // 64-col N half
  const int half = lane >> 4;       // lane half (ISA K-split)
  const int l    = lane & 15;
  const int rowA = mt * 16 + l;

  // A fragments for all 4 K-steps (16-bit A layout: lanes<16 K=0..7,16..23;
  // lanes>=16 K=8..15,24..31 relative to k0).
  V16u af[4];
#pragma unroll
  for (int ks = 0; ks < 4; ++ks) {
    int k0 = ks * 32;
    af[ks].h[0] = *(const v8bf*)&sA[rowA * 128 + k0 + 8 * half];
    af[ks].h[1] = *(const v8bf*)&sA[rowA * 128 + k0 + 16 + 8 * half];
  }

#pragma unroll
  for (int nt = 0; nt < 4; ++nt) {
    const int n = nh * 64 + nt * 16 + l;   // column this lane owns
    v8f acc = {};
#pragma unroll
    for (int ks = 0; ks < 4; ++ks) {
      // 16-bit B layout (matches A): lanes<16 K=0..7,16..23; lanes>=16 K=8..15,24..31.
      V16u bf;
      bf.h[0] = *(const v8bf*)&sBt[n * 128 + ks * 32 + 8 * half];
      bf.h[1] = *(const v8bf*)&sBt[n * 128 + ks * 32 + 16 + 8 * half];
      acc = __builtin_amdgcn_wmma_f32_16x16x32_bf16(
          /*neg_a=*/false, af[ks].v, /*neg_b=*/false, bf.v,
          /*c_mod=*/(short)0, acc, /*reuse_a=*/false, /*reuse_b=*/false);
    }
    // D layout: VGPR j -> row m0+mt*16+8*half+j, lane -> column n.
    const int rbase = m0 + mt * 16 + 8 * half;
    if (full) {
#pragma unroll
      for (int j = 0; j < 8; ++j) C[(rbase + j) * 128 + n] = acc[j];
    } else {
#pragma unroll
      for (int j = 0; j < 8; ++j) {
        int gr = rbase + j;
        if (gr < M) C[gr * 128 + n] = acc[j];
      }
    }
  }
}

// ---------------------------------------------------------------------------
// el[n,h] = <Wh[n, h*DH : (h+1)*DH], al[h]>, er likewise.
template <int H, int DH>
__global__ void attn_lr_k(const float* __restrict__ wh, const float* __restrict__ al,
                          const float* __restrict__ ar, float* __restrict__ el,
                          float* __restrict__ er) {
  int i = blockIdx.x * 256 + threadIdx.x;
  if (i >= NN * H) return;
  int n = i / H, h = i & (H - 1);
  const float* row = wh + n * DD + h * DH;
  const float* a = al + h * DH;
  const float* b = ar + h * DH;
  float sl = 0.f, sr = 0.f;
#pragma unroll 16
  for (int j = 0; j < DH; ++j) { float w = row[j]; sl += w * a[j]; sr += w * b[j]; }
  el[i] = sl;
  er[i] = sr;
}

// e = leaky_relu(el[src]+er[dst]); store e; segment-max over dst via atomicMax.
template <int H>
__global__ void edge_max_k(const float* __restrict__ el, const float* __restrict__ er,
                           const int* __restrict__ src, const int* __restrict__ dst,
                           float* __restrict__ ebuf, unsigned* __restrict__ mseg) {
  int i = blockIdx.x * 256 + threadIdx.x;
  if (i >= EE * H) return;
  int e = i / H, h = i & (H - 1);
  int s = src[e], d = dst[e];
  float v = el[s * H + h] + er[d * H + h];
  v = (v > 0.f) ? v : 0.2f * v;            // leaky_relu, slope 0.2
  ebuf[i] = v;
  atomicMax(&mseg[d * H + h], f2mono(v));
}

// ex = exp(e - m[dst]); in-place; denom[dst] += ex.
template <int H>
__global__ void edge_exp_k(float* __restrict__ ebuf, const int* __restrict__ dst,
                           const unsigned* __restrict__ mseg, float* __restrict__ denom) {
  int i = blockIdx.x * 256 + threadIdx.x;
  if (i >= EE * H) return;
  int e = i / H, h = i & (H - 1);
  int d = dst[e];
  float x = __expf(ebuf[i] - mono2f(mseg[d * H + h]));
  ebuf[i] = x;
  atomicAdd(&denom[d * H + h], x);
}

// agg[dst, d] += (ex/denom[dst]) * Wh[src, d]; thread per (edge, feature dim)
// -> coalesced gather of Wh[src] rows and coalesced atomic scatter.
template <int H, int DH>
__global__ void edge_agg_k(const float* __restrict__ ebuf, const float* __restrict__ denom,
                           const int* __restrict__ src, const int* __restrict__ dst,
                           const float* __restrict__ wh, float* __restrict__ agg) {
  int i = blockIdx.x * 256 + threadIdx.x;
  if (i >= EE * DD) return;
  int e = i >> 7;
  int dcol = i & (DD - 1);
  int h = dcol / DH;                        // DH is a power of two -> shift
  int s = src[e], d = dst[e];
  float alpha = ebuf[e * H + h] / denom[d * H + h];
  atomicAdd(&agg[d * DD + dcol], alpha * wh[s * DD + dcol]);
}

// out = agg + self_loop + bias (+relu); also emit bf16 copy for next layer.
__global__ void combine_k(const float* __restrict__ agg, const float* __restrict__ lo,
                          const float* __restrict__ bias, float* __restrict__ outF,
                          u16* __restrict__ outBF, int n, int relu) {
  int i = blockIdx.x * 256 + threadIdx.x;
  if (i >= n) return;
  float v = agg[i] + lo[i] + bias[i & (DD - 1)];
  if (relu) v = v > 0.f ? v : 0.f;
  if (outF) outF[i] = v;
  if (outBF) outBF[i] = f2bf(v);
}

// ---------------------------------------------------------------------------
extern "C" void kernel_launch(void* const* d_in, const int* in_sizes, int n_in,
                              void* d_out, int out_size, void* d_ws, size_t ws_size,
                              hipStream_t stream) {
  const float* h   = (const float*)d_in[0];
  const int*   src = (const int*)d_in[1];   // [R,E]
  const int*   dst = (const int*)d_in[2];   // [R,E]
  const float* W1  = (const float*)d_in[3];
  const float* al1 = (const float*)d_in[4];
  const float* ar1 = (const float*)d_in[5];
  const float* lw1 = (const float*)d_in[6];
  const float* b1  = (const float*)d_in[7];
  const float* W2  = (const float*)d_in[8];
  const float* al2 = (const float*)d_in[9];
  const float* ar2 = (const float*)d_in[10];
  const float* lw2 = (const float*)d_in[11];
  const float* b2  = (const float*)d_in[12];
  (void)in_sizes; (void)n_in; (void)out_size; (void)ws_size;

  // Workspace carve-out (~196 MB total).
  char* p = (char*)d_ws;
  auto take = [&](size_t bytes) {
    char* q = p;
    p += (bytes + 255) & ~(size_t)255;
    return q;
  };
  u16*      hbf   = (u16*)take((size_t)NN * DD * 2);            // layer input (bf16)
  u16*      wbf   = (u16*)take((size_t)(RR + 1) * DD * DD * 2); // W[r] + loop_w (bf16)
  float*    wh    = (float*)take((size_t)NN * DD * 4);          // per-relation Wh
  float*    lo    = (float*)take((size_t)NN * DD * 4);          // self-loop term
  float*    agg   = (float*)take((size_t)NN * DD * 4);          // cross-relation sum
  float*    el    = (float*)take((size_t)NN * 8 * 4);
  float*    er    = (float*)take((size_t)NN * 8 * 4);
  unsigned* mseg  = (unsigned*)take((size_t)NN * 8 * 4);        // encoded segment max
  float*    denom = (float*)take((size_t)NN * 8 * 4);
  float*    ebuf  = (float*)take((size_t)EE * 8 * 4);           // per-edge e / ex

  f32_to_bf16_k<<<CDIV(NN * DD, 256), 256, 0, stream>>>(h, hbf, NN * DD);

  for (int layer = 0; layer < 2; ++layer) {
    const float* W  = layer ? W2 : W1;
    const float* al = layer ? al2 : al1;
    const float* ar = layer ? ar2 : ar1;
    const float* lw = layer ? lw2 : lw1;
    const float* bs = layer ? b2 : b1;
    const int relu = layer ? 0 : 1;
    float* outF = layer ? (float*)d_out : nullptr;
    u16*   outB = layer ? nullptr : hbf;   // layer1 output feeds layer2 (in-place ok:
                                           // combine runs after all layer1 GEMMs)

    f32_to_bf16_k<<<CDIV(RR * DD * DD, 256), 256, 0, stream>>>(W, wbf, RR * DD * DD);
    f32_to_bf16_k<<<CDIV(DD * DD, 256), 256, 0, stream>>>(lw, wbf + RR * DD * DD, DD * DD);

    gemm_bf16_k128<<<CDIV(NN, 64), 256, 0, stream>>>(hbf, wbf + RR * DD * DD, lo, NN);
    hipMemsetAsync(agg, 0, (size_t)NN * DD * 4, stream);

    for (int r = 0; r < RR; ++r) {
      gemm_bf16_k128<<<CDIV(NN, 64), 256, 0, stream>>>(hbf, wbf + r * DD * DD, wh, NN);
      if (layer == 0) {
        constexpr int H = 8, DH = 16;
        attn_lr_k<H, DH><<<CDIV(NN * H, 256), 256, 0, stream>>>(
            wh, al + r * H * DH, ar + r * H * DH, el, er);
        fill_u32_k<<<CDIV(NN * H, 256), 256, 0, stream>>>(mseg, 0x007FFFFFu, NN * H);
        hipMemsetAsync(denom, 0, (size_t)NN * H * 4, stream);
        edge_max_k<H><<<CDIV(EE * H, 256), 256, 0, stream>>>(
            el, er, src + r * EE, dst + r * EE, ebuf, mseg);
        edge_exp_k<H><<<CDIV(EE * H, 256), 256, 0, stream>>>(ebuf, dst + r * EE, mseg, denom);
        edge_agg_k<H, DH><<<CDIV(EE * DD, 256), 256, 0, stream>>>(
            ebuf, denom, src + r * EE, dst + r * EE, wh, agg);
      } else {
        constexpr int H = 1, DH = 128;
        attn_lr_k<H, DH><<<CDIV(NN * H, 256), 256, 0, stream>>>(
            wh, al + r * H * DH, ar + r * H * DH, el, er);
        fill_u32_k<<<CDIV(NN * H, 256), 256, 0, stream>>>(mseg, 0x007FFFFFu, NN * H);
        hipMemsetAsync(denom, 0, (size_t)NN * H * 4, stream);
        edge_max_k<H><<<CDIV(EE * H, 256), 256, 0, stream>>>(
            el, er, src + r * EE, dst + r * EE, ebuf, mseg);
        edge_exp_k<H><<<CDIV(EE * H, 256), 256, 0, stream>>>(ebuf, dst + r * EE, mseg, denom);
        edge_agg_k<H, DH><<<CDIV(EE * DD, 256), 256, 0, stream>>>(
            ebuf, denom, src + r * EE, dst + r * EE, wh, agg);
      }
    }
    combine_k<<<CDIV(NN * DD, 256), 256, 0, stream>>>(agg, lo, bs, outF, outB, NN * DD, relu);
  }
}
